// RWKVTimeFirst_37864431681680
// MI455X (gfx1250) — matrix operand order repaired and checked by reading, not compiled, not run on hardware
//
#include <hip/hip_runtime.h>

// ---------------------------------------------------------------------------
// RWKV "time-first" block for MI455X (gfx1250, wave32, WMMA).
//
// Algebraic simplifications vs the reference:
//  * k_fin (scan carry of k) is dead code -> only k[:, -1, :] is needed.
//  * v_mix is constant along T -> o = v_fin @ Wo.T + bo computed once per
//    [B,C]; y = sigmoid(mix @ Wr.T) * o[b,c] broadcast.
// Big compute: two [16384,2048]x[2048,2048] GEMMs via
// v_wmma_f32_16x16x32_bf16 (bf16 inputs, f32 accumulate).
// Each wave computes a 32x64 output tile (8 accumulators): 12 b128 loads
// per 8 WMMAs -> high operand reuse for the matrix pipes.
// ---------------------------------------------------------------------------

typedef __bf16 bf16;
typedef __attribute__((ext_vector_type(16))) __bf16 v16bf;
typedef __attribute__((ext_vector_type(8)))  __bf16 v8bf;
typedef __attribute__((ext_vector_type(8)))  float  v8f;
typedef __attribute__((ext_vector_type(4)))  float  f32x4;
typedef __attribute__((ext_vector_type(4)))  unsigned int u32x4;

#define B_ 8
#define T_ 2048
#define C_ 2048
#define M_ (B_*T_)          // 16384 rows for the big GEMMs
#define LN_EPS 1e-12f

union FragAB { v16bf v; u32x4 q[2]; };

// ---------------------------------------------------------------------------
// 1) LayerNorm per (b,t) row; emit bf16 x_ln.
// ---------------------------------------------------------------------------
__global__ __launch_bounds__(256) void ln_kernel(const float* __restrict__ x,
                                                 const float* __restrict__ gamma,
                                                 const float* __restrict__ beta,
                                                 bf16* __restrict__ xln)
{
    const int row = blockIdx.x;          // b*T + t
    const int tid = threadIdx.x;
    const float* xr = x + (size_t)row * C_;

    float vals[8];
    f32x4 a = *(const f32x4*)(xr + tid * 8);
    f32x4 b = *(const f32x4*)(xr + tid * 8 + 4);
    vals[0]=a.x; vals[1]=a.y; vals[2]=a.z; vals[3]=a.w;
    vals[4]=b.x; vals[5]=b.y; vals[6]=b.z; vals[7]=b.w;

    float s = 0.f;
#pragma unroll
    for (int i = 0; i < 8; ++i) s += vals[i];

    __shared__ float red[256];
    red[tid] = s; __syncthreads();
    for (int st = 128; st > 0; st >>= 1) {
        if (tid < st) red[tid] += red[tid + st];
        __syncthreads();
    }
    const float mu = red[0] * (1.0f / (float)C_);
    __syncthreads();

    float sq = 0.f;
#pragma unroll
    for (int i = 0; i < 8; ++i) { float d = vals[i] - mu; sq += d * d; }
    red[tid] = sq; __syncthreads();
    for (int st = 128; st > 0; st >>= 1) {
        if (tid < st) red[tid] += red[tid + st];
        __syncthreads();
    }
    const float var  = red[0] * (1.0f / (float)C_);
    const float rstd = rsqrtf(var + LN_EPS);

    v8bf o;
#pragma unroll
    for (int i = 0; i < 8; ++i) {
        int c = tid * 8 + i;
        o[i] = (bf16)((vals[i] - mu) * rstd * gamma[c] + beta[c]);
    }
    *(v8bf*)(xln + (size_t)row * C_ + tid * 8) = o;
}

// ---------------------------------------------------------------------------
// 2) mix = x_ln * tf + shift(x_ln) * (1 - tf)   (bf16 in/out, f32 math)
// ---------------------------------------------------------------------------
__global__ void mix_kernel(const bf16* __restrict__ xln,
                           const float* __restrict__ tf,
                           bf16* __restrict__ mix)
{
    size_t idx    = (size_t)blockIdx.x * blockDim.x + threadIdx.x;
    const size_t total  = (size_t)M_ * C_;
    const size_t stride = (size_t)gridDim.x * blockDim.x;
    for (; idx < total; idx += stride) {
        const int    c   = (int)(idx & (C_ - 1));
        const size_t row = idx >> 11;            // / C_
        const int    t   = (int)(row & (T_ - 1));
        const float cur  = (float)xln[idx];
        const float prev = (t > 0) ? (float)xln[idx - C_] : 0.0f;
        const float f    = tf[c];
        mix[idx] = (bf16)(cur * f + prev * (1.0f - f));
    }
}

// ---------------------------------------------------------------------------
// 3) fp32 -> bf16 weight conversion
// ---------------------------------------------------------------------------
__global__ void cvt_bf16_kernel(const float* __restrict__ src,
                                bf16* __restrict__ dst, size_t n)
{
    size_t idx    = (size_t)blockIdx.x * blockDim.x + threadIdx.x;
    const size_t stride = (size_t)gridDim.x * blockDim.x;
    for (; idx < n; idx += stride) dst[idx] = (bf16)src[idx];
}

// ---------------------------------------------------------------------------
// 4) bf16 GEMM: out[m,n] = sum_k A[m,k] * W[n,k]   (A:[M,K], W:[N,K] row-major)
//    8 waves/block; each wave computes a 32x64 tile (2 M-frags x 4 N-frags,
//    8 accumulators); block tile = 128(M) x 128(N). Optional fused sigmoid.
// ---------------------------------------------------------------------------
__device__ __forceinline__ void store_acc(float* __restrict__ out, int Ndim,
                                          const v8f acc, int m0, int n0,
                                          int lane, int do_sig)
{
    const int laneHi = lane >> 4;
    const int lane16 = lane & 15;
#pragma unroll
    for (int i = 0; i < 8; ++i) {
        const int m = m0 + i + laneHi * 8;
        const int n = n0 + lane16;
        float v = acc[i];
        if (do_sig) v = 1.0f / (1.0f + __expf(-v));
        out[(size_t)m * Ndim + n] = v;
    }
}

__global__ __launch_bounds__(256) void gemm_bf16_kernel(
    const bf16* __restrict__ A,   // [M,K]
    const bf16* __restrict__ W,   // [N,K]
    float* __restrict__ out,      // [M,N]
    int Ndim, int Kdim, int do_sigmoid)
{
    const int lane   = threadIdx.x & 31;
    const int wave   = threadIdx.x >> 5;
    const int laneHi = lane >> 4;
    const int lane16 = lane & 15;
    const int wm     = wave & 3;     // 0..3 -> M sub-tile (32 rows each)
    const int wn     = wave >> 2;    // 0..1 -> N sub-tile (64 cols each)
    const int mBase  = blockIdx.y * 128 + wm * 32;
    const int nBase  = blockIdx.x * 128 + wn * 64;

    v8f acc[2][4];
#pragma unroll
    for (int i = 0; i < 2; ++i)
#pragma unroll
        for (int j = 0; j < 4; ++j) acc[i][j] = (v8f){};

    const bf16* Arow[2];
    const bf16* Wrow[4];
#pragma unroll
    for (int i = 0; i < 2; ++i)
        Arow[i] = A + (size_t)(mBase + 16 * i + lane16) * Kdim;
#pragma unroll
    for (int j = 0; j < 4; ++j)
        Wrow[j] = W + (size_t)(nBase + 16 * j + lane16) * Kdim;

    for (int kt = 0; kt < Kdim; kt += 32) {
        // A layout (16-bit A 16x32): lanes 0-15 hold K {0..7,16..23},
        // lanes 16-31 hold K {8..15,24..31}.
        const int ka = kt + (laneHi << 3);
        // B layout (16-bit B 32x16): lanes 0-15 hold K {0..7,16..23},
        // lanes 16-31 hold K {8..15,24..31}.
        const int kb = kt + (laneHi << 3);

        FragAB a[2], b[4];
#pragma unroll
        for (int i = 0; i < 2; ++i) {
            a[i].q[0] = *(const u32x4*)(Arow[i] + ka);
            a[i].q[1] = *(const u32x4*)(Arow[i] + ka + 16);
        }
#pragma unroll
        for (int j = 0; j < 4; ++j) {
            b[j].q[0] = *(const u32x4*)(Wrow[j] + kb);
            b[j].q[1] = *(const u32x4*)(Wrow[j] + kb + 16);
        }
#pragma unroll
        for (int i = 0; i < 2; ++i)
#pragma unroll
            for (int j = 0; j < 4; ++j)
                acc[i][j] = __builtin_amdgcn_wmma_f32_16x16x32_bf16(
                    false, a[i].v, false, b[j].v, (short)0, acc[i][j], false, false);
    }

#pragma unroll
    for (int i = 0; i < 2; ++i)
#pragma unroll
        for (int j = 0; j < 4; ++j)
            store_acc(out, Ndim, acc[i][j], mBase + 16 * i, nBase + 16 * j,
                      lane, do_sigmoid);
}

// ---------------------------------------------------------------------------
// 5) EMA scan over T per (b,c): v_fin = fold(cv*decay + v_t); also emits
//    new_v_state = v[:, -1, :]. Adjacent threads -> adjacent c: coalesced.
// ---------------------------------------------------------------------------
__global__ void scan_kernel(const float* __restrict__ v,
                            const float* __restrict__ td,
                            float* __restrict__ vfin,
                            float* __restrict__ vstate_out)
{
    const int id = blockIdx.x * blockDim.x + threadIdx.x;   // 0 .. B*C-1
    const int b  = id >> 11;          // / C_
    const int c  = id & (C_ - 1);
    const float decay = __expf(td[c]);
    const float* col = v + (size_t)b * T_ * C_ + c;
    float cv = 0.f, last = 0.f;
    for (int t = 0; t < T_; ++t) {
        last = col[(size_t)t * C_];
        cv   = cv * decay + last;
    }
    vfin[id]       = cv;
    vstate_out[id] = last;
}

// ---------------------------------------------------------------------------
// 6) new_k_state[b,c] = sum_j mix[b, T-1, j] * Wk[c, j]
// ---------------------------------------------------------------------------
__global__ void klast_kernel(const bf16* __restrict__ mix,
                             const float* __restrict__ Wk,
                             float* __restrict__ kout)
{
    const int id = blockIdx.x * blockDim.x + threadIdx.x;   // 0 .. B*C-1
    const int b  = id >> 11;
    const int c  = id & (C_ - 1);
    const bf16*  mrow = mix + (size_t)(b * T_ + (T_ - 1)) * C_;
    const float* wrow = Wk + (size_t)c * C_;
    float s = 0.f;
#pragma unroll 4
    for (int j = 0; j < C_; ++j) s += (float)mrow[j] * wrow[j];
    kout[id] = s;
}

// ---------------------------------------------------------------------------
// 7) o[b,c] = bo[c] + sum_j v_fin[b,j] * Wo[c,j]
// ---------------------------------------------------------------------------
__global__ void o_kernel(const float* __restrict__ vfin,
                         const float* __restrict__ Wo,
                         const float* __restrict__ bo,
                         float* __restrict__ o)
{
    const int id = blockIdx.x * blockDim.x + threadIdx.x;
    const int b  = id >> 11;
    const int c  = id & (C_ - 1);
    const float* vr = vfin + (size_t)b * C_;
    const float* wr = Wo + (size_t)c * C_;
    float s = bo[c];
#pragma unroll 4
    for (int j = 0; j < C_; ++j) s += vr[j] * wr[j];
    o[id] = s;
}

// ---------------------------------------------------------------------------
// 8) y[b,t,c] *= o[b,c]      (y already holds sigmoid(mix @ Wr.T))
// ---------------------------------------------------------------------------
__global__ void yfin_kernel(float* __restrict__ y, const float* __restrict__ o)
{
    size_t idx    = (size_t)blockIdx.x * blockDim.x + threadIdx.x;
    const size_t total  = (size_t)M_ * C_;
    const size_t stride = (size_t)gridDim.x * blockDim.x;
    for (; idx < total; idx += stride) {
        const int c = (int)(idx & (C_ - 1));
        const int b = (int)(idx >> 22);          // / (T_*C_)
        y[idx] *= o[b * C_ + c];
    }
}

// ---------------------------------------------------------------------------
extern "C" void kernel_launch(void* const* d_in, const int* in_sizes, int n_in,
                              void* d_out, int out_size, void* d_ws, size_t ws_size,
                              hipStream_t stream)
{
    (void)in_sizes; (void)n_in; (void)out_size; (void)ws_size;

    const float* x     = (const float*)d_in[0];
    const float* gamma = (const float*)d_in[1];
    const float* beta  = (const float*)d_in[2];
    const float* td    = (const float*)d_in[3];
    const float* tf    = (const float*)d_in[4];
    const float* Wk    = (const float*)d_in[5];
    const float* Wv    = (const float*)d_in[6];
    const float* Wr    = (const float*)d_in[7];
    const float* Wo    = (const float*)d_in[8];
    const float* bo    = (const float*)d_in[9];

    float* y      = (float*)d_out;                  // [B,T,C]
    float* kstate = y + (size_t)M_ * C_;            // [B,C]
    float* vstate = kstate + (size_t)B_ * C_;       // [B,C]

    char* ws = (char*)d_ws;
    bf16* xln  = (bf16*)ws;  ws += (size_t)M_ * C_ * sizeof(bf16);
    bf16* mixb = (bf16*)ws;  ws += (size_t)M_ * C_ * sizeof(bf16);
    bf16* Wvb  = (bf16*)ws;  ws += (size_t)C_ * C_ * sizeof(bf16);
    bf16* Wrb  = (bf16*)ws;  ws += (size_t)C_ * C_ * sizeof(bf16);
    float* vbuf = (float*)ws; ws += (size_t)M_ * C_ * sizeof(float);
    float* vfin = (float*)ws; ws += (size_t)B_ * C_ * sizeof(float);
    float* obuf = (float*)ws; ws += (size_t)B_ * C_ * sizeof(float);

    ln_kernel<<<M_, 256, 0, stream>>>(x, gamma, beta, xln);
    mix_kernel<<<8192, 256, 0, stream>>>(xln, tf, mixb);
    cvt_bf16_kernel<<<4096, 256, 0, stream>>>(Wv, Wvb, (size_t)C_ * C_);
    cvt_bf16_kernel<<<4096, 256, 0, stream>>>(Wr, Wrb, (size_t)C_ * C_);

    dim3 ggrid(C_ / 128, M_ / 128);   // (16, 128)
    gemm_bf16_kernel<<<ggrid, 256, 0, stream>>>(xln,  Wvb, vbuf, C_, C_, 0);
    gemm_bf16_kernel<<<ggrid, 256, 0, stream>>>(mixb, Wrb, y,    C_, C_, 1);

    scan_kernel<<<(B_ * C_) / 256, 256, 0, stream>>>(vbuf, td, vfin, vstate);
    klast_kernel<<<(B_ * C_) / 256, 256, 0, stream>>>(mixb, Wk, kstate);
    o_kernel<<<(B_ * C_) / 256, 256, 0, stream>>>(vfin, Wo, bo, obuf);
    yfin_kernel<<<8192, 256, 0, stream>>>(y, obuf);
}
